// MPNNRegressor_27986006901222
// MI455X (gfx1250) — matrix operand order repaired and checked by reading, not compiled, not run on hardware
//
#include <hip/hip_runtime.h>
#include <hip/hip_bf16.h>

// ---------------------------------------------------------------------------
// Problem constants (match reference)
// ---------------------------------------------------------------------------
#define NN_NODES 30000
#define NN_EDGES 120000
#define NN_GRAPHS 1000
#define DIM_IN 32
#define DIM_H 64
#define DIM_ED 8

typedef __attribute__((ext_vector_type(16))) __bf16 v16bf;
typedef __attribute__((ext_vector_type(8)))  float  v8f;
typedef unsigned int u32x4 __attribute__((ext_vector_type(4)));
typedef int          i32x4 __attribute__((ext_vector_type(4)));
typedef int          i32x8 __attribute__((ext_vector_type(8)));

union FragU { u32x4 u[2]; v16bf v; };

__device__ __forceinline__ unsigned short f2bfbits(float f) {
    unsigned u = __float_as_uint(f);
    unsigned r = 0x7FFFu + ((u >> 16) & 1u);   // round-to-nearest-even
    return (unsigned short)((u + r) >> 16);
}
__device__ __forceinline__ float bf2f(unsigned short b) {
    return __uint_as_float(((unsigned)b) << 16);
}

// ---------------------------------------------------------------------------
// TDM: issue a 2D tensor_load_to_lds (wave-level DMA, TENSORcnt-tracked).
// D# per CDNA5 ISA ch.8: group0 = {count/flags, lds_addr, global_addr, type},
// group1 = {mask/data_size, dims/tiles, dim0 stride}. data_size = 4 bytes.
// OOB rows (>= rows_avail) read as zero -> handles the M tail for free.
// ---------------------------------------------------------------------------
__device__ __forceinline__ void tdm_load_2d(const void* gptr, unsigned lds_off,
                                            unsigned rows_avail,   // tensor_dim1 (32b)
                                            unsigned row_dwords,   // tensor_dim0 == stride, dwords
                                            unsigned tile_rows,    // tile_dim1
                                            unsigned tile_dwords)  // tile_dim0
{
    unsigned long long ga = (unsigned long long)(uintptr_t)gptr;
    u32x4 g0 = { 1u,                                    // count=1, user mode
                 lds_off,                               // lds_addr (bytes)
                 (unsigned)(ga & 0xffffffffu),          // global_addr[31:0]
                 (unsigned)((ga >> 32) & 0x01ffffffu) | (2u << 30) }; // addr[56:32] | type=2
    i32x8 g1 = { (int)(2u << 16),                                   // data_size=4B, mask=0
                 (int)((row_dwords & 0xffffu) << 16),               // tensor_dim0[15:0]
                 (int)((row_dwords >> 16) | ((rows_avail & 0xffffu) << 16)), // dim0 hi | dim1 lo
                 (int)((rows_avail >> 16) | (tile_dwords << 16)),   // dim1 hi | tile_dim0
                 (int)(tile_rows & 0xffffu),                        // tile_dim1 (tile_dim2=0)
                 (int)row_dwords,                                   // dim0_stride[31:0]
                 0, 0 };                                            // stride hi / dim1_stride
    i32x4 gz = { 0, 0, 0, 0 };
#if defined(__clang_major__) && (__clang_major__ >= 23)
    i32x8 gz8 = { 0, 0, 0, 0, 0, 0, 0, 0 };
    __builtin_amdgcn_tensor_load_to_lds(g0, g1, gz, gz, gz8, 0);
#else
    __builtin_amdgcn_tensor_load_to_lds(g0, g1, gz, gz, 0);
#endif
}

// ---------------------------------------------------------------------------
// Zero-fill
// ---------------------------------------------------------------------------
__global__ void zero_f32(float* __restrict__ p, long long n) {
    long long i = (long long)blockIdx.x * blockDim.x + threadIdx.x;
    if (i < n) p[i] = 0.0f;
}

// ---------------------------------------------------------------------------
// ew2 (f32 [K][N]) -> bf16 transposed [N][K]
// ---------------------------------------------------------------------------
__global__ void transpose_bf16(const float* __restrict__ B,
                               unsigned short* __restrict__ Bt, int K2) {
    long long gid = (long long)blockIdx.x * blockDim.x + threadIdx.x;
    if (gid >= (long long)K2 * K2) return;
    int k = (int)(gid / K2);
    int n = (int)(gid % K2);
    Bt[(size_t)n * K2 + k] = f2bfbits(B[gid]);
}

// ---------------------------------------------------------------------------
// Edge MLP layer 1: h1[e,k] = relu(edge_attr[e,:8] @ ew1[:,k] + eb1[k]) (bf16)
// ---------------------------------------------------------------------------
__global__ void edge_hidden(const float* __restrict__ ea,
                            const float* __restrict__ ew1,
                            const float* __restrict__ eb1,
                            unsigned short* __restrict__ h1,
                            int E, int K2) {
    long long gid = (long long)blockIdx.x * blockDim.x + threadIdx.x;
    int perRow = K2 >> 3;
    int e = (int)(gid / perRow);
    if (e >= E) return;
    int kb = (int)(gid % perRow) << 3;

    float a[DIM_ED];
#pragma unroll
    for (int j = 0; j < DIM_ED; ++j) a[j] = ea[(size_t)e * DIM_ED + j];

    float acc[8];
#pragma unroll
    for (int o = 0; o < 8; ++o) acc[o] = eb1[kb + o];
#pragma unroll
    for (int j = 0; j < DIM_ED; ++j) {
        const float* wr = ew1 + (size_t)j * K2 + kb;
        float aj = a[j];
#pragma unroll
        for (int o = 0; o < 8; ++o) acc[o] += aj * wr[o];
    }
    unsigned int pk[4];
#pragma unroll
    for (int q = 0; q < 4; ++q) {
        unsigned lo = f2bfbits(fmaxf(acc[2 * q], 0.0f));
        unsigned hi = f2bfbits(fmaxf(acc[2 * q + 1], 0.0f));
        pk[q] = lo | (hi << 16);
    }
    u32x4 v = {pk[0], pk[1], pk[2], pk[3]};
    *(u32x4*)(h1 + (size_t)e * K2 + kb) = v;
}

// ---------------------------------------------------------------------------
// Big GEMM: C[M,N] = bf16( A[M,K](bf16) * B[K,N] + bias[N] ),
// B pre-transposed as Bt[N][K] bf16.
// 128x128 tile, BK=32, 8 waves x (32x64), v_wmma_f32_16x16x32_bf16.
// LDS tiles are filled by the Tensor Data Mover (tensor_load_to_lds),
// double-buffered, synchronized with s_wait_tensorcnt + workgroup barrier.
// ---------------------------------------------------------------------------
__global__ __launch_bounds__(256)
void gemm_wmma_bf16(const unsigned short* __restrict__ A,
                    const unsigned short* __restrict__ Bt,
                    const float* __restrict__ bias,
                    unsigned short* __restrict__ C,
                    int M, int K, int N) {
    __shared__ unsigned short As[2][128][32];
    __shared__ unsigned short Bs[2][128][32];

    const int tileN = blockIdx.x * 128;   // fast dim = N tiles -> A tile L2 reuse
    const int tileM = blockIdx.y * 128;
    const int tid  = threadIdx.x;
    const int lane = tid & 31;
    const int wave = tid >> 5;
    const int wm = (wave & 3) * 32;       // wave row offset in tile
    const int wn = (wave >> 2) * 64;      // wave col offset in tile

    v8f acc[2][4];
#pragma unroll
    for (int mi = 0; mi < 2; ++mi)
#pragma unroll
        for (int ni = 0; ni < 4; ++ni) acc[mi][ni] = (v8f)0.0f;

    const unsigned asOff = (unsigned)(uintptr_t)(&As[0][0][0]);
    const unsigned bsOff = (unsigned)(uintptr_t)(&Bs[0][0][0]);
    const unsigned rowDw = (unsigned)(K >> 1);          // dwords per row
    const unsigned aRows = (unsigned)(M - tileM);       // TDM zero-fills OOB rows
    const unsigned bRows = (unsigned)(N - tileN);
    const unsigned short* Abase = A  + (size_t)tileM * K;
    const unsigned short* Bbase = Bt + (size_t)tileN * K;

    auto issue = [&](int buf, int k0) {  // wave-uniform; called by wave 0 only
        tdm_load_2d(Abase + k0, asOff + (unsigned)buf * 8192u, aRows, rowDw, 128u, 16u);
        tdm_load_2d(Bbase + k0, bsOff + (unsigned)buf * 8192u, bRows, rowDw, 128u, 16u);
    };

    if (tid < 32) {
        issue(0, 0);
        __builtin_amdgcn_s_wait_tensorcnt((short)0);
    }
    __syncthreads();

    const int nk = K >> 5;
    const int ar  = wm + (lane & 15);
    const int akb = (lane & 16) ? 8 : 0;   // A frag K split {0-7,16-23}/{8-15,24-31}
    const int bkb = (lane & 16) ? 8 : 0;   // B frag K split {0-7,16-23}/{8-15,24-31}
    const int bc  = wn + (lane & 15);

    for (int kt = 0; kt < nk; ++kt) {
        const int cur = kt & 1;
        if ((kt + 1 < nk) && (tid < 32)) issue(cur ^ 1, (kt + 1) << 5);

        FragU af[2], bf[4];
#pragma unroll
        for (int mi = 0; mi < 2; ++mi) {
            af[mi].u[0] = *(const u32x4*)&As[cur][ar + 16 * mi][akb];
            af[mi].u[1] = *(const u32x4*)&As[cur][ar + 16 * mi][akb + 16];
        }
#pragma unroll
        for (int ni = 0; ni < 4; ++ni) {
            bf[ni].u[0] = *(const u32x4*)&Bs[cur][bc + 16 * ni][bkb];
            bf[ni].u[1] = *(const u32x4*)&Bs[cur][bc + 16 * ni][bkb + 16];
        }
#pragma unroll
        for (int mi = 0; mi < 2; ++mi)
#pragma unroll
            for (int ni = 0; ni < 4; ++ni)
                acc[mi][ni] = __builtin_amdgcn_wmma_f32_16x16x32_bf16(
                    false, af[mi].v, false, bf[ni].v,
                    (short)0, acc[mi][ni], false, false);

        if (tid < 32) __builtin_amdgcn_s_wait_tensorcnt((short)0);
        __syncthreads();
    }

    // epilogue: +bias, convert to bf16, store
    const int crow = (lane >> 4) << 3;  // 0 or 8
    const int ccol = lane & 15;
#pragma unroll
    for (int mi = 0; mi < 2; ++mi) {
#pragma unroll
        for (int ni = 0; ni < 4; ++ni) {
            const int gcol = tileN + wn + ni * 16 + ccol;
            const float bv = bias[gcol];
#pragma unroll
            for (int r = 0; r < 8; ++r) {
                const int grow = tileM + wm + mi * 16 + crow + r;
                if (grow < M)
                    C[(size_t)grow * N + gcol] = f2bfbits(acc[mi][ni][r] + bv);
            }
        }
    }
}

// ---------------------------------------------------------------------------
// Per-edge matvec + scatter-sum + degree: 64 lanes per edge (lane = output o)
// ---------------------------------------------------------------------------
__global__ void matvec_scatter(const unsigned short* __restrict__ w,
                               const float* __restrict__ hin,
                               const int* __restrict__ src,
                               const int* __restrict__ dst,
                               int din, int E,
                               float* __restrict__ s,
                               float* __restrict__ deg) {
    int t = blockIdx.x * 256 + threadIdx.x;
    int e = t >> 6, o = t & 63;
    if (e >= E) return;
    int sn = src[e], dn = dst[e];
    const unsigned short* we = w + (size_t)e * din * DIM_H + o;
    const float* xr = hin + (size_t)sn * din;
    float acc = 0.0f;
    for (int i = 0; i < din; ++i) acc += xr[i] * bf2f(we[i * DIM_H]);
    atomicAdd(&s[(size_t)dn * DIM_H + o], acc);
    if (o == 0) atomicAdd(&deg[dn], 1.0f);
}

// ---------------------------------------------------------------------------
// Node update: hout = relu(s/max(deg,1) + hin@root + cb)
// ---------------------------------------------------------------------------
__global__ void node_update(const float* __restrict__ s,
                            const float* __restrict__ deg,
                            const float* __restrict__ hin,
                            const float* __restrict__ root,
                            const float* __restrict__ cb,
                            int din, float* __restrict__ hout, int Nn) {
    int gid = blockIdx.x * 256 + threadIdx.x;
    if (gid >= Nn * DIM_H) return;
    int n = gid >> 6, o = gid & 63;
    float d = fmaxf(deg[n], 1.0f);
    float v = s[gid] / d + cb[o];
    const float* hr = hin + (size_t)n * din;
    for (int i = 0; i < din; ++i) v += hr[i] * root[i * DIM_H + o];
    hout[gid] = fmaxf(v, 0.0f);
}

// ---------------------------------------------------------------------------
// Global mean pool (sum + count)
// ---------------------------------------------------------------------------
__global__ void pool_sum(const float* __restrict__ h,
                         const int* __restrict__ batch,
                         float* __restrict__ pooled,
                         float* __restrict__ cnt, int Nn) {
    int gid = blockIdx.x * 256 + threadIdx.x;
    if (gid >= Nn * DIM_H) return;
    int n = gid >> 6, o = gid & 63;
    int g = batch[n];
    atomicAdd(&pooled[(size_t)g * DIM_H + o], h[gid]);
    if (o == 0) atomicAdd(&cnt[g], 1.0f);
}

// ---------------------------------------------------------------------------
// Readout MLP: out[g] = relu(gmean @ mw1 + mb1) @ mw2 + mb2
// ---------------------------------------------------------------------------
__global__ void readout(const float* __restrict__ pooled,
                        const float* __restrict__ cnt,
                        const float* __restrict__ mw1,
                        const float* __restrict__ mb1,
                        const float* __restrict__ mw2,
                        const float* __restrict__ mb2,
                        float* __restrict__ out) {
    __shared__ float gm[DIM_H];
    __shared__ float red[DIM_H];
    int g = blockIdx.x, j = threadIdx.x;
    float c = fmaxf(cnt[g], 1.0f);
    gm[j] = pooled[(size_t)g * DIM_H + j] / c;
    __syncthreads();
    float t = mb1[j];
    for (int i = 0; i < DIM_H; ++i) t += gm[i] * mw1[i * DIM_H + j];
    t = fmaxf(t, 0.0f);
    red[j] = t * mw2[j];
    __syncthreads();
    for (int ss = 32; ss > 0; ss >>= 1) {
        if (j < ss) red[j] += red[j + ss];
        __syncthreads();
    }
    if (j == 0) out[g] = red[0] + mb2[0];
}

// ---------------------------------------------------------------------------
// Host-side orchestration
// ---------------------------------------------------------------------------
extern "C" void kernel_launch(void* const* d_in, const int* in_sizes, int n_in,
                              void* d_out, int out_size, void* d_ws, size_t ws_size,
                              hipStream_t stream) {
    const float* x     = (const float*)d_in[0];
    const int*   eidx  = (const int*)d_in[1];
    const float* ea    = (const float*)d_in[2];
    const int*   batch = (const int*)d_in[3];
    const float* P[22];
    for (int i = 4; i < 26; ++i) P[i - 4] = (const float*)d_in[i];
    const float* mw1 = P[18]; const float* mb1 = P[19];
    const float* mw2 = P[20]; const float* mb2 = P[21];

    const int N = NN_NODES, E = NN_EDGES, G = NN_GRAPHS;
    const int* src = eidx;
    const int* dst = eidx + E;

    // workspace carve-up
    char* ws = (char*)d_ws;
    size_t off = 0;
    auto carve = [&](size_t bytes) { char* p = ws + off; off = (off + bytes + 255) & ~(size_t)255; return p; };
    unsigned short* h_edge = (unsigned short*)carve((size_t)E * 4096 * 2);
    unsigned short* w_edge = (unsigned short*)carve((size_t)E * 4096 * 2);
    unsigned short* ew2t   = (unsigned short*)carve((size_t)4096 * 4096 * 2);
    float* hA     = (float*)carve((size_t)N * DIM_H * 4);
    float* hB     = (float*)carve((size_t)N * DIM_H * 4);
    float* sbuf   = (float*)carve((size_t)N * DIM_H * 4);
    float* deg    = (float*)carve((size_t)N * 4);
    float* pooled = (float*)carve((size_t)G * DIM_H * 4);
    float* cnt    = (float*)carve((size_t)G * 4);
    (void)ws_size; (void)in_sizes; (void)n_in;

    auto run_layer = [&](const float* ew1, const float* eb1, const float* ew2,
                         const float* eb2, const float* root, const float* cb,
                         const float* hin, float* hout, int din) {
        const int K2 = din * DIM_H;
        long long tT = (long long)K2 * K2;
        transpose_bf16<<<(unsigned)((tT + 255) / 256), 256, 0, stream>>>(ew2, ew2t, K2);
        long long tE = (long long)E * (K2 >> 3);
        edge_hidden<<<(unsigned)((tE + 255) / 256), 256, 0, stream>>>(ea, ew1, eb1, h_edge, E, K2);
        dim3 gg(K2 / 128, (E + 127) / 128);
        gemm_wmma_bf16<<<gg, 256, 0, stream>>>(h_edge, ew2t, eb2, w_edge, E, K2, K2);
        zero_f32<<<(N * DIM_H + 255) / 256, 256, 0, stream>>>(sbuf, (long long)N * DIM_H);
        zero_f32<<<(N + 255) / 256, 256, 0, stream>>>(deg, (long long)N);
        matvec_scatter<<<(E * DIM_H + 255) / 256, 256, 0, stream>>>(w_edge, hin, src, dst, din, E, sbuf, deg);
        node_update<<<(N * DIM_H + 255) / 256, 256, 0, stream>>>(sbuf, deg, hin, root, cb, din, hout, N);
    };

    run_layer(P[0],  P[1],  P[2],  P[3],  P[4],  P[5],  x,  hA, DIM_IN);
    run_layer(P[6],  P[7],  P[8],  P[9],  P[10], P[11], hA, hB, DIM_H);
    run_layer(P[12], P[13], P[14], P[15], P[16], P[17], hB, hA, DIM_H);

    zero_f32<<<(G * DIM_H + 255) / 256, 256, 0, stream>>>(pooled, (long long)G * DIM_H);
    zero_f32<<<(G + 255) / 256, 256, 0, stream>>>(cnt, (long long)G);
    pool_sum<<<(N * DIM_H + 255) / 256, 256, 0, stream>>>(hA, batch, pooled, cnt, N);
    readout<<<G, DIM_H, 0, stream>>>(pooled, cnt, mw1, mb1, mw2, mb2, (float*)d_out);
    (void)out_size;
}
